// NeRFRenderer_2259152798108
// MI455X (gfx1250) — matrix operand order repaired and checked by reading, not compiled, not run on hardware
//
#include <hip/hip_runtime.h>
#include <math.h>

typedef __attribute__((ext_vector_type(16))) _Float16 v16h;
typedef __attribute__((ext_vector_type(8)))  float    v8f;

#define NUM_RAYS   16384
#define NSTEPS     128
#define HID        64
#define WPB        4            // waves per block
#define TPB        (WPB * 32)
#define NBLOCKS    512          // 2048 waves -> 8 rays per wave, uniform

// LDS fragment buffer layout (24 fragments, per-lane swizzled, 1KB each):
//  [0..3]   dW1^T   (M-tiles, K padded 3->32)
//  [4..7]   cW1b^T  (M-tiles, K padded 16->32)
//  [8..15]  cW2^T   (nt*2+kc)
//  [16..23] cW3^T   (nt*2+kc)
#define F_DW1  0
#define F_CW1  4
#define F_CW2  8
#define F_CW3  16
#define NFRAG  24

union V16 { v16h v; unsigned w[8]; };

__device__ __forceinline__ void wsync() {
  asm volatile("" ::: "memory");
  __builtin_amdgcn_wave_barrier();
}

__device__ __forceinline__ v8f wmma_f16(v16h a, v16h b, v8f c) {
  return __builtin_amdgcn_wmma_f32_16x16x32_f16(false, a, false, b, (short)0, c, false, false);
}

__device__ __forceinline__ unsigned pack2(float a, float b) {
  union { _Float16 h[2]; unsigned u; } t;
  t.h[0] = (_Float16)a; t.h[1] = (_Float16)b;
  return t.u;
}

// Build a 16x32 f16 A-fragment of W^T, i.e. A[m][k] = W[k][m], from row-major
// f32 W[Ktot][ld].  Tile origin (k0, m0); out-of-range K/M -> 0 (padding).
// A layout: M = lane%16; lanes 0-15 hold K = k0+{0..15},
// lanes 16-31 hold K = k0+{16..31}, 2 halves per VGPR.
__device__ __forceinline__ v16h load_aT(const float* __restrict__ W, int ld,
                                        int k0, int m0, int Ktot, int Mtot,
                                        int lane) {
  int m  = m0 + (lane & 15);
  int kb = k0 + ((lane & 16) ? 16 : 0);
  v16h f;
#pragma unroll
  for (int i = 0; i < 8; ++i) {
    int k = kb + i;
    f[i] = (k < Ktot && m < Mtot) ? (_Float16)W[k * ld + m] : (_Float16)0.0f;
  }
#pragma unroll
  for (int i = 0; i < 8; ++i) {
    int k = kb + 8 + i;
    f[8 + i] = (k < Ktot && m < Mtot) ? (_Float16)W[k * ld + m] : (_Float16)0.0f;
  }
  return f;
}

// Per-lane pre-swizzled fragment read: 32B contiguous -> 2x ds_load_b128.
__device__ __forceinline__ v16h lda_lds(const _Float16* __restrict__ buf,
                                        int frag, int lane) {
  return *(const v16h*)(buf + (frag * 32 + lane) * 16);
}

struct BPair { v16h b0, b1; };

// Convert four C tiles (output features mt*16.., samples in lanes) + per-feature
// add[] (64 floats, LDS) with optional relu into the two B-fragments of the
// 64-feature transposed activation.  Cross-half move is a single lane-16 swap.
__device__ __forceinline__ BPair c_to_b(const v8f* c, const float* __restrict__ add,
                                        bool relu, int lane) {
  const bool lo = (lane & 16) == 0;
  const int  mb = lo ? 0 : 8;
  unsigned pk[4][4];
#pragma unroll
  for (int mt = 0; mt < 4; ++mt) {
    const float* bp = add + mt * 16 + mb;
#pragma unroll
    for (int j = 0; j < 4; ++j) {
      float a = c[mt][2 * j]     + bp[2 * j];
      float b = c[mt][2 * j + 1] + bp[2 * j + 1];
      if (relu) { a = fmaxf(a, 0.0f); b = fmaxf(b, 0.0f); }
      pk[mt][j] = pack2(a, b);
    }
  }
  V16 f0, f1;
#pragma unroll
  for (int j = 0; j < 4; ++j) {
    unsigned s0 = (unsigned)__shfl_xor((int)pk[0][j], 16, 32);
    unsigned s1 = (unsigned)__shfl_xor((int)pk[1][j], 16, 32);
    unsigned s2 = (unsigned)__shfl_xor((int)pk[2][j], 16, 32);
    unsigned s3 = (unsigned)__shfl_xor((int)pk[3][j], 16, 32);
    f0.w[j]     = lo ? pk[0][j] : s1;
    f0.w[4 + j] = lo ? s0       : pk[1][j];
    f1.w[j]     = lo ? pk[2][j] : s3;
    f1.w[4 + j] = lo ? s2       : pk[3][j];
  }
  BPair r; r.b0 = f0.v; r.b1 = f1.v;
  return r;
}

// Single 16-feature C tile (density_out^T) -> one B fragment, K padded 16->32.
__device__ __forceinline__ v16h cd_to_b(v8f cd, const float* __restrict__ bsel,
                                        int lane) {
  const bool lo = (lane & 16) == 0;
  unsigned pk[4];
#pragma unroll
  for (int j = 0; j < 4; ++j)
    pk[j] = pack2(cd[2 * j] + bsel[2 * j], cd[2 * j + 1] + bsel[2 * j + 1]);
  V16 f;
#pragma unroll
  for (int j = 0; j < 4; ++j) {
    unsigned s = (unsigned)__shfl_xor((int)pk[j], 16, 32);
    f.w[j]     = lo ? pk[j] : 0u;
    f.w[4 + j] = lo ? s     : 0u;
  }
  return f.v;
}

__global__ void __launch_bounds__(TPB, 4)
nerf_fused_kernel(const float* __restrict__ rays_o, const float* __restrict__ rays_d,
                  const float* __restrict__ dW1, const float* __restrict__ db1,
                  const float* __restrict__ dW2, const float* __restrict__ db2,
                  const float* __restrict__ cW1, const float* __restrict__ cb1,
                  const float* __restrict__ cW2, const float* __restrict__ cb2,
                  const float* __restrict__ cW3, const float* __restrict__ cb3,
                  const float* __restrict__ cW4, const float* __restrict__ cb4,
                  float* __restrict__ out) {
  __shared__ _Float16 s_frag[NFRAG * 32 * 16];   // 24KB, shared by all waves
  __shared__ float    s_db1[HID];
  __shared__ float    s_cb2[HID];
  __shared__ float    s_cb3[HID];
  __shared__ float    s_dirpart[WPB][HID];       // per-ray sh @ cW1a + cb1

  const int tid  = threadIdx.x;
  const int lane = tid & 31;
  const int wid  = tid >> 5;
  const bool lo  = (lane & 16) == 0;

  // ---- one-time: pre-swizzle weight fragments into LDS (wave 0, serialized) ----
  if (wid == 0) {
#pragma unroll 1
    for (int f = 0; f < 4; ++f) {
      v16h x = load_aT(dW1, HID, 0, f * 16, 3, 64, lane);
      *(v16h*)(s_frag + ((F_DW1 + f) * 32 + lane) * 16) = x;
    }
#pragma unroll 1
    for (int f = 0; f < 4; ++f) {
      v16h x = load_aT(cW1 + 16 * HID, HID, 0, f * 16, 16, 64, lane);
      *(v16h*)(s_frag + ((F_CW1 + f) * 32 + lane) * 16) = x;
    }
#pragma unroll 1
    for (int f = 0; f < 8; ++f) {
      v16h x = load_aT(cW2, HID, (f & 1) * 32, (f >> 1) * 16, 64, 64, lane);
      *(v16h*)(s_frag + ((F_CW2 + f) * 32 + lane) * 16) = x;
    }
#pragma unroll 1
    for (int f = 0; f < 8; ++f) {
      v16h x = load_aT(cW3, HID, (f & 1) * 32, (f >> 1) * 16, 64, 64, lane);
      *(v16h*)(s_frag + ((F_CW3 + f) * 32 + lane) * 16) = x;
    }
  }
  for (int i = tid; i < HID; i += TPB) {
    s_db1[i] = db1[i];
    s_cb2[i] = cb2[i];
    s_cb3[i] = cb3[i];
  }
  __syncthreads();

  // ---- small weight fragments resident in registers (4 frags = 32 VGPRs) ----
  v16h AdW2[2], AcW4[2];
#pragma unroll 1
  for (int kc = 0; kc < 2; ++kc) {
    AdW2[kc] = load_aT(dW2, 16, kc * 32, 0, 64, 16, lane);
    AcW4[kc] = load_aT(cW4, 3, kc * 32, 0, 64, 3, lane);   // M padded 3->16
  }

  float db2sel[8];
#pragma unroll
  for (int v = 0; v < 8; ++v) db2sel[v] = db2[(lo ? 0 : 8) + v];
  const float c40 = cb4[0], c41 = cb4[1], c42 = cb4[2];

  const int gwave  = blockIdx.x * WPB + wid;
  const int nwaves = NBLOCKS * WPB;

#pragma unroll 1
  for (int ray = gwave; ray < NUM_RAYS; ray += nwaves) {
    // ---- per-ray setup (all lanes redundantly) ----
    float ox = rays_o[ray * 3 + 0], oy = rays_o[ray * 3 + 1], oz = rays_o[ray * 3 + 2];
    float dx = rays_d[ray * 3 + 0], dy = rays_d[ray * 3 + 1], dz = rays_d[ray * 3 + 2];

    float ix = 1.0f / ((fabsf(dx) < 1e-15f) ? 1e-15f : dx);
    float iy = 1.0f / ((fabsf(dy) < 1e-15f) ? 1e-15f : dy);
    float iz = 1.0f / ((fabsf(dz) < 1e-15f) ? 1e-15f : dz);
    float t1x = (-1.0f - ox) * ix, t2x = (1.0f - ox) * ix;
    float t1y = (-1.0f - oy) * iy, t2y = (1.0f - oy) * iy;
    float t1z = (-1.0f - oz) * iz, t2z = (1.0f - oz) * iz;
    float nearv = fmaxf(fmaxf(fminf(t1x, t2x), fminf(t1y, t2y)), fminf(t1z, t2z));
    float farv  = fminf(fminf(fmaxf(t1x, t2x), fmaxf(t1y, t2y)), fmaxf(t1z, t2z));
    nearv = fmaxf(nearv, 0.05f);
    farv  = fmaxf(farv, nearv + 1e-6f);
    float span = farv - nearv;
    float dlt_mid  = span * (1.0f / (NSTEPS - 1));
    float dlt_last = span * (1.0f / NSTEPS);

    // ---- per-ray: dirpart[j] = sum_k sh[k]*cW1a[k][j] + cb1[j] ----
    {
      float inv = rsqrtf(dx * dx + dy * dy + dz * dz);
      float sx = dx * inv, sy = dy * inv, sz = dz * inv;
      float x2 = sx * sx, y2 = sy * sy, z2 = sz * sz;
      float xy = sx * sy, yz = sy * sz, xz = sx * sz;
      float sh[16];
      sh[0]  = 0.28209479177387814f;
      sh[1]  = -0.4886025119029199f * sy;
      sh[2]  =  0.4886025119029199f * sz;
      sh[3]  = -0.4886025119029199f * sx;
      sh[4]  =  1.0925484305920792f * xy;
      sh[5]  = -1.0925484305920792f * yz;
      sh[6]  =  0.31539156525252005f * (2.0f * z2 - x2 - y2);
      sh[7]  = -1.0925484305920792f * xz;
      sh[8]  =  0.5462742152960396f * (x2 - y2);
      sh[9]  = -0.5900435899266435f * sy * (3.0f * x2 - y2);
      sh[10] =  2.890611442640554f  * xy * sz;
      sh[11] = -0.4570457994644658f * sy * (4.0f * z2 - x2 - y2);
      sh[12] =  0.3731763325901154f * sz * (2.0f * z2 - 3.0f * x2 - 3.0f * y2);
      sh[13] = -0.4570457994644658f * sx * (4.0f * z2 - x2 - y2);
      sh[14] =  1.445305721320277f  * sz * (x2 - y2);
      sh[15] = -0.5900435899266435f * sx * (x2 - 3.0f * y2);
#pragma unroll
      for (int rep = 0; rep < 2; ++rep) {
        int j = lane + rep * 32;
        float acc = cb1[j];
#pragma unroll
        for (int k = 0; k < 16; ++k) acc += sh[k] * cW1[k * HID + j];
        s_dirpart[wid][j] = acc;
      }
    }
    wsync();

    float T = 1.0f, wsum = 0.0f, accR = 0.0f, accG = 0.0f, accB = 0.0f;

#pragma unroll 1
    for (int t = 0; t < NSTEPS / 16; ++t) {
      // Launder the fragment base through an opaque register so LICM cannot
      // hoist the (loop-invariant) LDS fragment loads out of the tile loop —
      // that hoisting materializes 24 fragments in VGPRs and spills to scratch.
      unsigned fzero = 0;
      asm volatile("" : "+v"(fzero));
      const _Float16* fb = s_frag + fzero;

      // ---- per-lane sample coords (sample = lane%16) ----
      int sl = lane & 15;
      int gi = t * 16 + sl;
      float zv = nearv + span * ((float)gi * (1.0f / (NSTEPS - 1)));
      float px = fminf(fmaxf(ox + dx * zv, -1.0f), 1.0f);
      float py = fminf(fmaxf(oy + dy * zv, -1.0f), 1.0f);
      float pz = fminf(fmaxf(oz + dz * zv, -1.0f), 1.0f);

      // ---- xyz^T B-fragment (K padded 3->32; lanes>=16 are pad rows) ----
      V16 bx;
#pragma unroll
      for (int j = 0; j < 8; ++j) bx.w[j] = 0u;
      bx.w[0] = lo ? pack2(px, py)   : 0u;
      bx.w[1] = lo ? pack2(pz, 0.0f) : 0u;

      // ---- density layer 1: h1^T = dW1^T @ xyz^T (4 WMMAs) ----
      v8f ct[4];
#pragma unroll
      for (int mt = 0; mt < 4; ++mt) {
        v8f a = {};
        ct[mt] = wmma_f16(lda_lds(fb, F_DW1 + mt, lane), bx.v, a);
      }
      BPair Bh1 = c_to_b(ct, s_db1, true, lane);

      // ---- density layer 2: density_out^T = dW2^T @ h1^T (2 WMMAs) ----
      v8f cd = {};
      cd = wmma_f16(AdW2[0], Bh1.b0, cd);
      cd = wmma_f16(AdW2[1], Bh1.b1, cd);
      float densraw = cd[0] + db2sel[0];      // feature 0, valid on lanes 0..15
      v16h Bd = cd_to_b(cd, db2sel, lane);

      // ---- color layer 1: cW1b^T @ density_out^T + dirpart (4 WMMAs) ----
#pragma unroll
      for (int mt = 0; mt < 4; ++mt) {
        v8f a = {};
        ct[mt] = wmma_f16(lda_lds(fb, F_CW1 + mt, lane), Bd, a);
      }
      BPair Bc1 = c_to_b(ct, s_dirpart[wid], true, lane);

      // ---- color layer 2 (8 WMMAs, weights streamed from LDS) ----
#pragma unroll
      for (int nt = 0; nt < 4; ++nt) {
        v8f acc = {};
        acc = wmma_f16(lda_lds(fb, F_CW2 + nt * 2 + 0, lane), Bc1.b0, acc);
        acc = wmma_f16(lda_lds(fb, F_CW2 + nt * 2 + 1, lane), Bc1.b1, acc);
        ct[nt] = acc;
      }
      BPair Bc2 = c_to_b(ct, s_cb2, true, lane);

      // ---- color layer 3 (8 WMMAs) ----
#pragma unroll
      for (int nt = 0; nt < 4; ++nt) {
        v8f acc = {};
        acc = wmma_f16(lda_lds(fb, F_CW3 + nt * 2 + 0, lane), Bc2.b0, acc);
        acc = wmma_f16(lda_lds(fb, F_CW3 + nt * 2 + 1, lane), Bc2.b1, acc);
        ct[nt] = acc;
      }
      BPair Bc3 = c_to_b(ct, s_cb3, true, lane);

      // ---- color layer 4 + sigmoid (2 WMMAs; rgb = features 0..2, lanes 0..15) ----
      v8f cc = {};
      cc = wmma_f16(AcW4[0], Bc3.b0, cc);
      cc = wmma_f16(AcW4[1], Bc3.b1, cc);
      float r = 1.0f / (1.0f + expf(-(cc[0] + c40)));
      float g = 1.0f / (1.0f + expf(-(cc[1] + c41)));
      float b = 1.0f / (1.0f + expf(-(cc[2] + c42)));

      // ---- composite 16 samples (lane = sample; product scan via shfl) ----
      float dens  = expf(densraw);
      float delta = (gi == NSTEPS - 1) ? dlt_last : dlt_mid;
      float alpha = 1.0f - expf(-delta * dens);
      float p = 1.0f - alpha + 1e-15f;

      float cum = p;
#pragma unroll
      for (int off = 1; off < 16; off <<= 1) {
        float u = __shfl_up(cum, off, 16);
        cum = (sl >= off) ? cum * u : cum;
      }
      float excl = __shfl_up(cum, 1, 16);
      excl = (sl == 0) ? 1.0f : excl;
      float w = T * excl * alpha;

      bool m = (w > 1e-4f);
      float lw = lo ? w : 0.0f;
      float lr = (lo && m) ? w * r : 0.0f;
      float lg = (lo && m) ? w * g : 0.0f;
      float lb = (lo && m) ? w * b : 0.0f;
#pragma unroll
      for (int off = 16; off >= 1; off >>= 1) {
        lw += __shfl_xor(lw, off);
        lr += __shfl_xor(lr, off);
        lg += __shfl_xor(lg, off);
        lb += __shfl_xor(lb, off);
      }
      wsum += lw; accR += lr; accG += lg; accB += lb;
      T *= __shfl(cum, 15);   // lane 15 holds the whole-tile product
    }

    if (lane == 0) {
      float bg = 1.0f - wsum;
      out[ray * 3 + 0] = accR + bg;
      out[ray * 3 + 1] = accG + bg;
      out[ray * 3 + 2] = accB + bg;
    }
  }
}

extern "C" void kernel_launch(void* const* d_in, const int* in_sizes, int n_in,
                              void* d_out, int out_size, void* d_ws, size_t ws_size,
                              hipStream_t stream) {
  (void)in_sizes; (void)n_in; (void)d_ws; (void)ws_size; (void)out_size;
  const float* rays_o = (const float*)d_in[0];
  const float* rays_d = (const float*)d_in[1];
  const float* dW1 = (const float*)d_in[2];
  const float* db1 = (const float*)d_in[3];
  const float* dW2 = (const float*)d_in[4];
  const float* db2 = (const float*)d_in[5];
  const float* cW1 = (const float*)d_in[6];
  const float* cb1 = (const float*)d_in[7];
  const float* cW2 = (const float*)d_in[8];
  const float* cb2 = (const float*)d_in[9];
  const float* cW3 = (const float*)d_in[10];
  const float* cb3 = (const float*)d_in[11];
  const float* cW4 = (const float*)d_in[12];
  const float* cb4 = (const float*)d_in[13];
  float* out = (float*)d_out;

  hipLaunchKernelGGL(nerf_fused_kernel, dim3(NBLOCKS), dim3(TPB), 0, stream,
                     rays_o, rays_d, dW1, db1, dW2, db2, cW1, cb1,
                     cW2, cb2, cW3, cb3, cW4, cb4, out);
}
